// MetaBEVWithModalFusion_88424786690138
// MI455X (gfx1250) — matrix operand order repaired and checked by reading, not compiled, not run on hardware
//
#include <hip/hip_runtime.h>
#include <math.h>

// ---------------------------------------------------------------------------
// MetaBEV modal fusion for gfx1250 (MI455X).
// All GEMM-shaped math runs on v_wmma_f32_16x16x32_bf16 (fp32 accumulate).
// Layouts follow CDNA5 ISA 7.12.2. K-loops are ping-pong software pipelined
// (no register-swap moves); GEMM A-tiles staged to LDS via
// global_load_async_to_lds_b128 (+ s_wait_asynccnt) when available.
// ---------------------------------------------------------------------------

typedef __bf16 bfv8  __attribute__((ext_vector_type(8)));
typedef __bf16 bfv16 __attribute__((ext_vector_type(16)));
typedef float  fv8   __attribute__((ext_vector_type(8)));
typedef int    v4i   __attribute__((ext_vector_type(4)));
typedef __attribute__((address_space(1))) v4i* glb_v4i_ptr;  // global (AS1)
typedef __attribute__((address_space(3))) v4i* lds_v4i_ptr;  // LDS (AS3)

#define WMMA_BF16(a, b, c) \
    __builtin_amdgcn_wmma_f32_16x16x32_bf16(false, (a), false, (b), (short)0, (c), false, false)

#if defined(__AMDGCN__) && __has_builtin(__builtin_amdgcn_global_load_async_to_lds_b128) && \
    __has_builtin(__builtin_amdgcn_s_wait_asynccnt)
#define USE_ASYNC_LDS 1
#else
#define USE_ASYNC_LDS 0
#endif

__device__ __forceinline__ bfv16 load_frag_A(const __bf16* base, int ld, int m0, int k0, int lane) {
    int row = m0 + (lane & 15);
    int lh  = lane >> 4;
    const __bf16* p = base + (size_t)row * ld + k0 + 8 * lh;
    bfv8 c0 = *(const bfv8*)(p);
    bfv8 c1 = *(const bfv8*)(p + 16);
    return __builtin_shufflevector(c0, c1, 0,1,2,3,4,5,6,7,8,9,10,11,12,13,14,15);
}

__device__ __forceinline__ bfv16 load_frag_A_lds(const __bf16* sA, int k0, int lane) {
    const __bf16* p = sA + (lane & 15) * 256 + k0 + 8 * (lane >> 4);
    bfv8 c0 = *(const bfv8*)(p);
    bfv8 c1 = *(const bfv8*)(p + 16);
    return __builtin_shufflevector(c0, c1, 0,1,2,3,4,5,6,7,8,9,10,11,12,13,14,15);
}

__device__ __forceinline__ bfv16 load_frag_B(const __bf16* base, int ld, int n0, int k0, int lane) {
    int row = n0 + (lane & 15);
    int lh  = lane >> 4;
    const __bf16* p = base + (size_t)row * ld + k0 + 8 * lh;
    bfv8 c0 = *(const bfv8*)(p);
    bfv8 c1 = *(const bfv8*)(p + 16);
    return __builtin_shufflevector(c0, c1, 0,1,2,3,4,5,6,7,8,9,10,11,12,13,14,15);
}

// Generic C[M,N] = A[M,256] * B[N,256]^T (+bias) with epilogue modes. K == 256.
// mode 0: bf16 store C[m][n]            mode 1: bf16 store transposed C[n][m]
// mode 2: f32 C[m][n] += v              mode 3: f32 C[m][n] = v
// mode 4: f32 C[m][n] += gate[m*8+e]*v  mode 5: f32 store transposed C[n][m]
__global__ __launch_bounds__(128)
void gemm_wmma(const __bf16* __restrict__ A, int lda,
               const __bf16* __restrict__ B, int ldb,
               const float* __restrict__ bias,
               void* __restrict__ Cp, int ldc,
               int M, int N, int mode, float alpha,
               const float* __restrict__ gate, int expert) {
    __shared__ __bf16 sA[16 * 256];
    int tid = threadIdx.x, wv = tid >> 5, lane = tid & 31;
    int m0  = blockIdx.y * 16;
    int n0w = blockIdx.x * 256 + wv * 64;

    // Stage the 16 x 256 A tile to LDS once per workgroup (shared by 4 waves).
    {
        for (int c = tid; c < 512; c += 128) {  // 512 16B chunks
            int row = c >> 5, col8 = (c & 31) * 8;
            const __bf16* g = A + (size_t)(m0 + row) * lda + col8;
            __bf16* l = sA + row * 256 + col8;
#if USE_ASYNC_LDS
            __builtin_amdgcn_global_load_async_to_lds_b128(
                (glb_v4i_ptr)(unsigned long long)g,
                (lds_v4i_ptr)(unsigned)(unsigned long long)l,
                0, 0);
#else
            *(bfv8*)l = *(const bfv8*)g;
#endif
        }
#if USE_ASYNC_LDS
        __builtin_amdgcn_s_wait_asynccnt(0);
#endif
        __syncthreads();
    }
    if (n0w >= N || m0 >= M) return;

    fv8 acc[4] = {};
    bfv16 a[2];
    bfv16 b[2][4];
    a[0] = load_frag_A_lds(sA, 0, lane);
#pragma unroll
    for (int t = 0; t < 4; t++) b[0][t] = load_frag_B(B, ldb, n0w + t * 16, 0, lane);
#pragma unroll
    for (int k0 = 0; k0 < 256; k0 += 32) {   // fully unrolled: ping-pong, no copies
        const int st = (k0 >> 5) & 1;
        if (k0 + 32 < 256) {
            a[st ^ 1] = load_frag_A_lds(sA, k0 + 32, lane);
#pragma unroll
            for (int t = 0; t < 4; t++) b[st ^ 1][t] = load_frag_B(B, ldb, n0w + t * 16, k0 + 32, lane);
        }
#pragma unroll
        for (int t = 0; t < 4; t++) acc[t] = WMMA_BF16(a[st], b[st][t], acc[t]);
    }

    int lh = lane >> 4, nl = lane & 15;
#pragma unroll
    for (int t = 0; t < 4; t++) {
        int n = n0w + t * 16 + nl;
        float bv = bias ? bias[n] : 0.0f;
        if (mode == 0) {
            __bf16* C = (__bf16*)Cp;
            for (int r = 0; r < 8; r++)
                C[(size_t)(m0 + 8 * lh + r) * ldc + n] = (__bf16)(acc[t][r] * alpha + bv);
        } else if (mode == 1) {
            __bf16* C = (__bf16*)Cp;
            bfv8 pk;
            for (int r = 0; r < 8; r++) pk[r] = (__bf16)(acc[t][r] * alpha + bv);
            *(bfv8*)(C + (size_t)n * ldc + m0 + 8 * lh) = pk;
        } else if (mode == 2) {
            float* C = (float*)Cp;
            for (int r = 0; r < 8; r++)
                C[(size_t)(m0 + 8 * lh + r) * ldc + n] += acc[t][r] * alpha + bv;
        } else if (mode == 3) {
            float* C = (float*)Cp;
            for (int r = 0; r < 8; r++)
                C[(size_t)(m0 + 8 * lh + r) * ldc + n] = acc[t][r] * alpha + bv;
        } else if (mode == 4) {
            float* C = (float*)Cp;
            for (int r = 0; r < 8; r++) {
                int m = m0 + 8 * lh + r;
                C[(size_t)m * ldc + n] += gate[(size_t)m * 8 + expert] * acc[t][r];
            }
        } else {
            float* C = (float*)Cp;
            for (int r = 0; r < 8; r++)
                C[(size_t)n * ldc + m0 + 8 * lh + r] = acc[t][r] * alpha + bv;
        }
    }
}

// scores[z][q][k] = alpha * Q[qoff+q][32h..] . K[koff+k][32h..]
__global__ __launch_bounds__(128)
void attn_scores(const __bf16* __restrict__ Q, const __bf16* __restrict__ K,
                 float* __restrict__ S, int Sq, int Sk, int qoff, int koff,
                 int heads, int head_base, float alpha) {
    int tid = threadIdx.x, wv = tid >> 5, lane = tid & 31;
    int z = blockIdx.z;
    int h = head_base + (z % heads);
    int m0 = blockIdx.y * 16;
    int n0w = blockIdx.x * 256 + wv * 64;
    if (n0w >= Sk) return;
    const __bf16* Qb = Q + (size_t)qoff * 256;
    const __bf16* Kb = K + (size_t)koff * 256;
    int kch = h * 32;
    bfv16 a = load_frag_A(Qb, 256, m0, kch, lane);
    bfv16 b[4];
#pragma unroll
    for (int t = 0; t < 4; t++) b[t] = load_frag_B(Kb, 256, n0w + t * 16, kch, lane);
    fv8 acc[4] = {};
#pragma unroll
    for (int t = 0; t < 4; t++) acc[t] = WMMA_BF16(a, b[t], acc[t]);
    float* Sz = S + (size_t)z * Sq * Sk;
    int lh = lane >> 4, nl = lane & 15;
#pragma unroll
    for (int t = 0; t < 4; t++) {
        int n = n0w + t * 16 + nl;
        for (int r = 0; r < 8; r++)
            Sz[(size_t)(m0 + 8 * lh + r) * Sk + n] = acc[t][r] * alpha;
    }
}

// Row softmax f32 -> bf16.
__global__ __launch_bounds__(256)
void softmax_bf16(const float* __restrict__ S, __bf16* __restrict__ P, int L) {
    int row = blockIdx.x, tid = threadIdx.x;
    const float* x = S + (size_t)row * L;
    __bf16* y = P + (size_t)row * L;
    __shared__ float red[256];
    float mx = -1e30f;
    for (int i = tid; i < L; i += 256) mx = fmaxf(mx, x[i]);
    red[tid] = mx; __syncthreads();
    for (int s = 128; s > 0; s >>= 1) { if (tid < s) red[tid] = fmaxf(red[tid], red[tid + s]); __syncthreads(); }
    mx = red[0]; __syncthreads();
    float sum = 0.0f;
    for (int i = tid; i < L; i += 256) sum += __expf(x[i] - mx);
    red[tid] = sum; __syncthreads();
    for (int s = 128; s > 0; s >>= 1) { if (tid < s) red[tid] += red[tid + s]; __syncthreads(); }
    float inv = 1.0f / red[0];
    for (int i = tid; i < L; i += 256) y[i] = (__bf16)(__expf(x[i] - mx) * inv);
}

// O[qoff+q][32h + d] = sum_k P[z][q][k] * Vt[32h+d][koff+k]
// Each wave: 32 query rows x 32 head channels; 2-stage ping-pong over K
// (Sk is a multiple of 64), no register-swap copies.
__global__ __launch_bounds__(128)
void attn_out(const __bf16* __restrict__ P, const __bf16* __restrict__ Vt,
              __bf16* __restrict__ O, int Sq, int Sk, int qoff, int koff, int T,
              int heads, int head_base) {
    int tid = threadIdx.x, wv = tid >> 5, lane = tid & 31;
    int z = blockIdx.z;
    int h = head_base + (z % heads);
    int m0 = (blockIdx.y * 4 + wv) * 32;
    if (m0 >= Sq) return;
    const __bf16* Pz = P + (size_t)z * Sq * Sk;
    const __bf16* Vb = Vt + koff;
    fv8 acc[2][2] = {};
    bfv16 a[2][2], b[2][2];
    a[0][0] = load_frag_A(Pz, Sk, m0, 0, lane);
    a[0][1] = load_frag_A(Pz, Sk, m0 + 16, 0, lane);
    b[0][0] = load_frag_B(Vb, T, h * 32, 0, lane);
    b[0][1] = load_frag_B(Vb, T, h * 32 + 16, 0, lane);
    for (int k0 = 0; k0 < Sk; k0 += 64) {
        // stage 1 fragments at k0+32 (always in range: Sk % 64 == 0)
        a[1][0] = load_frag_A(Pz, Sk, m0, k0 + 32, lane);
        a[1][1] = load_frag_A(Pz, Sk, m0 + 16, k0 + 32, lane);
        b[1][0] = load_frag_B(Vb, T, h * 32, k0 + 32, lane);
        b[1][1] = load_frag_B(Vb, T, h * 32 + 16, k0 + 32, lane);
#pragma unroll
        for (int mt = 0; mt < 2; mt++)
#pragma unroll
            for (int nt = 0; nt < 2; nt++)
                acc[mt][nt] = WMMA_BF16(a[0][mt], b[0][nt], acc[mt][nt]);
        if (k0 + 64 < Sk) {  // stage 0 fragments for next iteration
            a[0][0] = load_frag_A(Pz, Sk, m0, k0 + 64, lane);
            a[0][1] = load_frag_A(Pz, Sk, m0 + 16, k0 + 64, lane);
            b[0][0] = load_frag_B(Vb, T, h * 32, k0 + 64, lane);
            b[0][1] = load_frag_B(Vb, T, h * 32 + 16, k0 + 64, lane);
        }
#pragma unroll
        for (int mt = 0; mt < 2; mt++)
#pragma unroll
            for (int nt = 0; nt < 2; nt++)
                acc[mt][nt] = WMMA_BF16(a[1][mt], b[1][nt], acc[mt][nt]);
    }
    int lh = lane >> 4, nl = lane & 15;
#pragma unroll
    for (int mt = 0; mt < 2; mt++)
        for (int r = 0; r < 8; r++) {
            size_t m = (size_t)(qoff + m0 + mt * 16 + 8 * lh + r);
            O[m * 256 + h * 32 + nl]      = (__bf16)acc[mt][0][r];
            O[m * 256 + h * 32 + 16 + nl] = (__bf16)acc[mt][1][r];
        }
}

// [C,H,W] inputs -> token-major bf16 blocks (+meta average).
__global__ void pack_kernel(const float* __restrict__ D, const float* __restrict__ L,
                            const float* __restrict__ Ev, __bf16* __restrict__ meta,
                            __bf16* __restrict__ d16, __bf16* __restrict__ l16,
                            __bf16* __restrict__ e16) {
    int idx = blockIdx.x * blockDim.x + threadIdx.x;  // T*E
    int c = idx & 255, t = idx >> 8;
    int blk = t >> 10, s = t & 1023;
    int h = (blk >> 1) * 32 + (s >> 5);
    int w = (blk & 1) * 32 + (s & 31);
    size_t src = (size_t)c * 4096 + h * 64 + w;
    float dv = D[src], lv = L[src], ev = Ev[src];
    d16[idx] = (__bf16)dv; l16[idx] = (__bf16)lv; e16[idx] = (__bf16)ev;
    meta[idx] = (__bf16)((dv + lv + ev) * (1.0f / 3.0f));
}

__global__ void f32_to_bf16(const float* __restrict__ s, __bf16* __restrict__ d, int n) {
    int i = blockIdx.x * blockDim.x + threadIdx.x;
    if (i < n) d[i] = (__bf16)s[i];
}

__global__ void zero_f32(float* __restrict__ p, int n) {
    int i = blockIdx.x * blockDim.x + threadIdx.x;
    if (i < n) p[i] = 0.0f;
}

// gate[t][e] = softmax_e(fused[t] . Wg[e] + bg[e])
__global__ __launch_bounds__(256)
void gate_kernel(const float* __restrict__ fused, const float* __restrict__ Wg,
                 const float* __restrict__ bg, float* __restrict__ gate) {
    int lt = threadIdx.x >> 3;
    int e  = threadIdx.x & 7;
    int t  = blockIdx.x * 32 + lt;
    float d = bg[e];
    const float* x = fused + (size_t)t * 256;
    const float* w = Wg + (size_t)e * 256;
    for (int c = 0; c < 256; c++) d += x[c] * w[c];
    __shared__ float sh[32][8];
    sh[lt][e] = d; __syncthreads();
    float mx = sh[lt][0];
    for (int i = 1; i < 8; i++) mx = fmaxf(mx, sh[lt][i]);
    float sum = 0.0f;
    for (int i = 0; i < 8; i++) sum += __expf(sh[lt][i] - mx);
    gate[(size_t)t * 8 + e] = __expf(d - mx) / sum;
}

// moe blocks -> raster-order sequence (+ gate@be bias), bf16.
__global__ void blocks_to_seq(const float* __restrict__ moe, const float* __restrict__ gate,
                              const float* __restrict__ be, __bf16* __restrict__ seq) {
    int idx = blockIdx.x * blockDim.x + threadIdx.x;  // T*E
    int c = idx & 255, tok2 = idx >> 8;
    int hh = tok2 >> 6, ww = tok2 & 63;
    int blk = (hh >> 5) * 2 + (ww >> 5);
    int s = (hh & 31) * 32 + (ww & 31);
    int t = blk * 1024 + s;
    float v = moe[(size_t)t * 256 + c];
    const float* g = gate + (size_t)t * 8;
    for (int e = 0; e < 8; e++) v += g[e] * be[(size_t)e * 256 + c];
    seq[idx] = (__bf16)v;
}

extern "C" void kernel_launch(void* const* d_in, const int* in_sizes, int n_in,
                              void* d_out, int out_size, void* d_ws, size_t ws_size,
                              hipStream_t stream) {
    const int T = 4096, E = 256, HEADS = 8;
    const float alpha = 0.17677669529663687f;  // 1/sqrt(32)
    char* ws = (char*)d_ws;
    size_t off = 0;
    auto take = [&](size_t bytes) { size_t o = off; off = (off + bytes + 255) & ~(size_t)255; return (void*)(ws + o); };

    __bf16* meta16 = (__bf16*)take((size_t)T * E * 2);
    __bf16* mod16[3];
    for (int i = 0; i < 3; i++) mod16[i] = (__bf16*)take((size_t)T * E * 2);
    __bf16* Wi16[4]; for (int i = 0; i < 4; i++) Wi16[i] = (__bf16*)take((size_t)3 * E * E * 2);
    __bf16* Wo16[4]; for (int i = 0; i < 4; i++) Wo16[i] = (__bf16*)take((size_t)E * E * 2);
    __bf16* We16  = (__bf16*)take((size_t)8 * E * E * 2);
    __bf16* Q16   = (__bf16*)take((size_t)T * E * 2);
    __bf16* K16   = (__bf16*)take((size_t)T * E * 2);
    __bf16* Vt16  = (__bf16*)take((size_t)E * T * 2);   // [256][T]
    __bf16* O16   = (__bf16*)take((size_t)T * E * 2);
    float*  fusedf = (float*)take((size_t)T * E * 4);
    __bf16* fused16 = (__bf16*)take((size_t)T * E * 2);
    float*  gateb  = (float*)take((size_t)T * 8 * 4);
    float*  moef   = (float*)take((size_t)T * E * 4);
    __bf16* seq16  = (__bf16*)take((size_t)T * E * 2);
    float*  Sbuf   = (float*)take((size_t)4096 * 4096 * 4);   // 64MB (>= 8*1024*1024)
    __bf16* Pbuf   = (__bf16*)take((size_t)4096 * 4096 * 2);  // 32MB
    (void)ws_size; (void)in_sizes; (void)n_in; (void)out_size;

    auto conv = [&](int idx, __bf16* dst, int n) {
        f32_to_bf16<<<(n + 255) / 256, 256, 0, stream>>>((const float*)d_in[idx], dst, n);
    };
    const int wiIdx[4] = {3, 7, 11, 15}, woIdx[4] = {5, 9, 13, 17};
    for (int i = 0; i < 4; i++) { conv(wiIdx[i], Wi16[i], 3 * E * E); conv(woIdx[i], Wo16[i], E * E); }
    conv(21, We16, 8 * E * E);

    pack_kernel<<<(T * E) / 256, 256, 0, stream>>>(
        (const float*)d_in[0], (const float*)d_in[1], (const float*)d_in[2],
        meta16, mod16[0], mod16[1], mod16[2]);

    // --- cross attention: 3 modalities ---
    for (int mi = 0; mi < 3; mi++) {
        const float* bi = (const float*)d_in[wiIdx[mi] + 1];
        const float* bo = (const float*)d_in[woIdx[mi] + 1];
        gemm_wmma<<<dim3(1, T / 16), 128, 0, stream>>>(meta16, E, Wi16[mi], E, bi,
                                                       Q16, E, T, E, 0, 1.0f, nullptr, 0);
        gemm_wmma<<<dim3(1, T / 16), 128, 0, stream>>>(mod16[mi], E, Wi16[mi] + (size_t)E * E, E, bi + E,
                                                       K16, E, T, E, 0, 1.0f, nullptr, 0);
        gemm_wmma<<<dim3(1, T / 16), 128, 0, stream>>>(mod16[mi], E, Wi16[mi] + (size_t)2 * E * E, E, bi + 2 * E,
                                                       Vt16, T, T, E, 1, 1.0f, nullptr, 0);
        for (int blk = 0; blk < 4; blk++) {
            int o = blk * 1024;
            attn_scores<<<dim3(4, 64, HEADS), 128, 0, stream>>>(Q16, K16, Sbuf, 1024, 1024, o, o, HEADS, 0, alpha);
            softmax_bf16<<<HEADS * 1024, 256, 0, stream>>>(Sbuf, Pbuf, 1024);
            attn_out<<<dim3(1, 8, HEADS), 128, 0, stream>>>(Pbuf, Vt16, O16, 1024, 1024, o, o, T, HEADS, 0);
        }
        gemm_wmma<<<dim3(1, T / 16), 128, 0, stream>>>(O16, E, Wo16[mi], E, bo,
                                                       fusedf, E, T, E, (mi == 0 ? 3 : 2), 1.0f, nullptr, 0);
    }

    // --- MoE ---
    gate_kernel<<<T / 32, 256, 0, stream>>>(fusedf, (const float*)d_in[19], (const float*)d_in[20], gateb);
    f32_to_bf16<<<(T * E) / 256, 256, 0, stream>>>(fusedf, fused16, T * E);
    zero_f32<<<(T * E) / 256, 256, 0, stream>>>(moef, T * E);
    for (int e = 0; e < 8; e++)
        gemm_wmma<<<dim3(1, T / 16), 128, 0, stream>>>(fused16, E, We16 + (size_t)e * E * E, E, nullptr,
                                                       moef, E, T, E, 4, 1.0f, gateb, e);
    blocks_to_seq<<<(T * E) / 256, 256, 0, stream>>>(moef, gateb, (const float*)d_in[22], seq16);

    // --- final global self-attention ---
    const float* bim = (const float*)d_in[16];
    const float* bom = (const float*)d_in[18];
    gemm_wmma<<<dim3(1, T / 16), 128, 0, stream>>>(seq16, E, Wi16[3], E, bim,
                                                   Q16, E, T, E, 0, 1.0f, nullptr, 0);
    gemm_wmma<<<dim3(1, T / 16), 128, 0, stream>>>(seq16, E, Wi16[3] + (size_t)E * E, E, bim + E,
                                                   K16, E, T, E, 0, 1.0f, nullptr, 0);
    gemm_wmma<<<dim3(1, T / 16), 128, 0, stream>>>(seq16, E, Wi16[3] + (size_t)2 * E * E, E, bim + 2 * E,
                                                   Vt16, T, T, E, 1, 1.0f, nullptr, 0);
    for (int h = 0; h < HEADS; h++) {
        attn_scores<<<dim3(16, 256, 1), 128, 0, stream>>>(Q16, K16, Sbuf, 4096, 4096, 0, 0, 1, h, alpha);
        softmax_bf16<<<4096, 256, 0, stream>>>(Sbuf, Pbuf, 4096);
        attn_out<<<dim3(1, 32, 1), 128, 0, stream>>>(Pbuf, Vt16, O16, 4096, 4096, 0, 0, T, 1, h);
    }
    gemm_wmma<<<dim3(1, T / 16), 128, 0, stream>>>(O16, E, Wo16[3], E, bom,
                                                   d_out, 4096, T, E, 5, 1.0f, nullptr, 0);
}
